// KAN1_23751169147141
// MI455X (gfx1250) — matrix operand order repaired and checked by reading, not compiled, not run on hardware
//
#include <hip/hip_runtime.h>
#include <hip/hip_bf16.h>

typedef __attribute__((ext_vector_type(16))) _Float16 v16h;
typedef __attribute__((ext_vector_type(8)))  float    v8f;
typedef __attribute__((ext_vector_type(4)))  unsigned int v4u;
typedef __attribute__((ext_vector_type(8)))  int      v8i;
typedef __attribute__((ext_vector_type(4)))  int      v4i;

#define B_  64
#define P_  196
#define C_  768
#define HS_ 384
#define M_  (B_*C_)   // 49152 rows (b*C + c)

// ---------------------------------------------------------------------------
// TDM: async 2D tile load Global -> LDS (CDNA5 tensor_load_to_lds).
// D# per CDNA5 ISA (6-arg builtin on this toolchain).
// ---------------------------------------------------------------------------
__device__ inline void tdm_load_2d(unsigned lds_off, const float* gptr,
                                   unsigned tile_d0, unsigned tile_d1,
                                   unsigned tdim0, unsigned tdim1,
                                   unsigned stride0, unsigned pad_code)
{
  unsigned long long ga = (unsigned long long)(const void*)gptr;
  v4u g0;
  g0.x = 1u;                                   // count = 1 valid descriptor
  g0.y = lds_off;                              // LDS byte address
  g0.z = (unsigned)ga;                         // global addr [31:0]
  g0.w = (unsigned)((ga >> 32) & 0x01ffffffu) | (2u << 30);  // [56:32] | type=2
  v8i g1;
  g1[0] = (int)((2u << 16) | (1u << 20) | (pad_code << 22)); // 4B, pad_en, interval
  g1[1] = (int)((tdim0 & 0xffffu) << 16);
  g1[2] = (int)((tdim0 >> 16) | ((tdim1 & 0xffffu) << 16));
  g1[3] = (int)((tdim1 >> 16) | (tile_d0 << 16));
  g1[4] = (int)tile_d1;
  g1[5] = (int)stride0;
  g1[6] = 0;
  g1[7] = 0;
  v4i z4 = {0, 0, 0, 0};
  v8i z8 = {0, 0, 0, 0, 0, 0, 0, 0};
  __builtin_amdgcn_tensor_load_to_lds(g0, g1, z4, z4, z8, 0);
}

// ---------------------------------------------------------------------------
// Pack [spline_w^T ; base_w^T] (f32) into f16 WMMA B-fragment order.
// ---------------------------------------------------------------------------
__device__ inline unsigned int pack2h(float a, float b) {
  union { _Float16 h[2]; unsigned int u; } p;
  p.h[0] = (_Float16)a; p.h[1] = (_Float16)b;
  return p.u;
}

__global__ void kan_pack_w(const float* __restrict__ spline_w,
                           const float* __restrict__ base_w,
                           unsigned int* __restrict__ out,
                           int N, int din, int NBP, int KCS, int KCT)
{
  int t = blockIdx.x * 256 + threadIdx.x;
  int total = KCT * NBP * 256;
  if (t >= total) return;
  int dw   = t & 7;
  int lane = (t >> 3) & 31;
  int nb   = (t >> 8) % NBP;
  int kc   = (t >> 8) / NBP;
  int n    = nb * 16 + (lane & 15);
  int hi   = lane >> 4;
  int k0   = hi * 8 + 2 * dw + ((dw >= 4) ? 8 : 0);
  float w0 = 0.f, w1 = 0.f;
  if (n < N) {
    if (kc < KCS) {                     // spline region: k = j*8 + f
      int kg = kc * 32 + k0;
      const float* row = spline_w + (size_t)n * (size_t)(din * 8);
      w0 = row[kg]; w1 = row[kg + 1];
    } else {                            // base (silu) region, zero padded
      int j = (kc - KCS) * 32 + k0;
      const float* row = base_w + (size_t)n * (size_t)din;
      if (j     < din) w0 = row[j];
      if (j + 1 < din) w1 = row[j + 1];
    }
  }
  out[t] = pack2h(w0, w1);
}

// ---------------------------------------------------------------------------
// Fused KAN-linear GEMM with TDM double-buffered LDS staging.
// ---------------------------------------------------------------------------
template<int LAYER>
__global__ __launch_bounds__(128)
void kan_gemm(const float* __restrict__ src,
              const unsigned int* __restrict__ wpack,
              const float* __restrict__ bias,
              const float* __restrict__ xskip,
              float* __restrict__ dst)
{
  constexpr int DIN   = (LAYER == 1) ? P_  : HS_;
  constexpr int NBP   = (LAYER == 1) ? 24  : 16;
  constexpr int KCS   = DIN / 4;                   // spline K chunks
  constexpr int NSLAB = (DIN + 31) / 32;           // slabs == base K chunks
  constexpr int SLABF = (LAYER == 1) ? 32 * 65 : 64 * 33;  // floats per buffer
  constexpr int SMEMF = (2 * SLABF > 64 * 65) ? 2 * SLABF : 64 * 65;

  __shared__ float smem[SMEMF];

  const int tid  = threadIdx.x;
  const int lane = tid & 31;
  const int wv   = tid >> 5;
  const int hi   = lane >> 4;
  const int rowl = lane & 15;
  const int rs   = wv * 16 + rowl;     // row within 64-row tile

  const int m0   = blockIdx.x * 64;    // never straddles b (64 | 768)
  const int nbg0 = blockIdx.y * 4;
  const int bB   = m0 / C_;
  const int c0   = m0 % C_;

  const v8f vzero = {0.f,0.f,0.f,0.f,0.f,0.f,0.f,0.f};
  v8f acc[4];
  #pragma unroll
  for (int i = 0; i < 4; ++i) acc[i] = vzero;

  // per-lane base into the packed weights; per-chunk advance = NBP*256 uints
  const unsigned int* wlane = wpack + (size_t)(nbg0 * 256) + (size_t)lane * 8;

  auto issue_slab = [&](int s, int buf) {
    unsigned lds_off = (unsigned)(unsigned long long)(const void*)(smem + buf * SLABF);
    const int js0 = s * 32;
    if (LAYER == 1) {
      // tile [c=64 fast][p=32], stride C; pad 1 dword per 64 -> [j][r] stride 65
      const float* g = src + (size_t)bB * (P_ * C_) + (size_t)js0 * C_ + c0;
      tdm_load_2d(lds_off, g, 64, 32, 64, (unsigned)(DIN - js0), C_, 5);
    } else {
      // tile [j=32 fast][r=64], stride HS; pad 1 dword per 32 -> [r][j] stride 33
      const float* g = src + (size_t)m0 * HS_ + js0;
      tdm_load_2d(lds_off, g, 32, 64, 32, 64, HS_, 4);
    }
  };

  auto sread = [&](const float* sb, int j) -> float {
    return (LAYER == 1) ? sb[j * 65 + rs] : sb[rs * 33 + j];
  };

  auto wmma4 = [&](const v16h& a, const v16h& b0, const v16h& b1,
                   const v16h& b2, const v16h& b3) {
    acc[0] = __builtin_amdgcn_wmma_f32_16x16x32_f16(false, a, false, b0, (short)0, acc[0], false, false);
    acc[1] = __builtin_amdgcn_wmma_f32_16x16x32_f16(false, a, false, b1, (short)0, acc[1], false, false);
    acc[2] = __builtin_amdgcn_wmma_f32_16x16x32_f16(false, a, false, b2, (short)0, acc[2], false, false);
    acc[3] = __builtin_amdgcn_wmma_f32_16x16x32_f16(false, a, false, b3, (short)0, acc[3], false, false);
  };

  // whole compute for one slab; `guard` is constant at every call site
  auto compute_slab = [&](int s, int cur, bool guard) {
    const float* sb = smem + cur * SLABF;
    const int js0 = s * 32;

    // ---- spline chunks: kc covers elements j = 4kc..4kc+3
    const int kcb = s * 8;
    const int kce = (kcb + 8 < KCS) ? (kcb + 8) : KCS;
    for (int kc = kcb; kc < kce; ++kc) {
      const unsigned int* wk = wlane + (size_t)kc * (NBP * 256);
      const v16h bf0 = *(const v16h*)(wk);
      const v16h bf1 = *(const v16h*)(wk + 256);
      const v16h bf2 = *(const v16h*)(wk + 512);
      const v16h bf3 = *(const v16h*)(wk + 768);

      const int jb = kc * 4 - js0;
      // A: lane<16 -> K 0..7 (j=jb), 16..23 (j=jb+2); lane>=16 -> +1 element
      const float xa = sread(sb, jb + hi);
      const float xb = sread(sb, jb + 2 + hi);
      v16h a;
      #pragma unroll
      for (int f = 0; f < 8; ++f) {
        const float g  = -1.0f + (float)f * (2.0f / 7.0f);
        const float ta = (xa - g) * 3.5f;
        const float tb = (xb - g) * 3.5f;
        a[f]     = (_Float16)__expf(-ta * ta);
        a[f + 8] = (_Float16)__expf(-tb * tb);
      }
      wmma4(a, bf0, bf1, bf2, bf3);
    }

    // ---- base (silu) chunk: kc = KCS + s, K-local == j-local
    {
      const int kc = KCS + s;
      const unsigned int* wk = wlane + (size_t)kc * (NBP * 256);
      const v16h bf0 = *(const v16h*)(wk);
      const v16h bf1 = *(const v16h*)(wk + 256);
      const v16h bf2 = *(const v16h*)(wk + 512);
      const v16h bf3 = *(const v16h*)(wk + 768);
      v16h a;
      #pragma unroll
      for (int h = 0; h < 16; ++h) {
        const int kl = h + ((h >= 8) ? 8 : 0) + hi * 8;
        float xv = sread(sb, kl);
        if (guard && (js0 + kl >= DIN)) xv = 0.f;   // only last layer-1 slab
        const float sg = __builtin_amdgcn_rcpf(1.0f + __expf(-xv));
        a[h] = (_Float16)(xv * sg);
      }
      wmma4(a, bf0, bf1, bf2, bf3);
    }
  };

  // prologue: slab 0 into buffer 0
  if (wv == 0) {
    issue_slab(0, 0);
    __builtin_amdgcn_s_wait_tensorcnt(0);
  }
  __syncthreads();

  // main loop: all but the last slab, branch-free body, async prefetch ahead
  int cur = 0;
  for (int s = 0; s < NSLAB - 1; ++s) {
    if (wv == 0) issue_slab(s + 1, cur ^ 1);
    compute_slab(s, cur, false);
    if (wv == 0) __builtin_amdgcn_s_wait_tensorcnt(0);  // prefetch landed
    __syncthreads();                                    // all reads of cur done
    cur ^= 1;
  }
  // peeled last slab: guard is compile-time constant (only layer 1 has a tail)
  compute_slab(NSLAB - 1, cur, LAYER == 1);
  __syncthreads();   // slab buffers free before epilogue reuses smem

  // ---------------- epilogue: transpose through LDS, b128 stores
  if (LAYER == 1) {
    // tile [m 64][n 64] stride 65; then float4 rows of h[m][nbg0*16 + n]
    #pragma unroll
    for (int nb = 0; nb < 4; ++nb)
      #pragma unroll
      for (int r = 0; r < 8; ++r)
        smem[(wv * 16 + hi * 8 + r) * 65 + nb * 16 + rowl] = acc[nb][r];
    __syncthreads();
    for (int q = tid; q < 1024; q += 128) {
      const int mloc = q >> 4;
      const int n4   = (q & 15) * 4;
      const float4 bb = *(const float4*)(bias + nbg0 * 16 + n4);
      float4 v;
      v.x = smem[mloc * 65 + n4 + 0] + bb.x;
      v.y = smem[mloc * 65 + n4 + 1] + bb.y;
      v.z = smem[mloc * 65 + n4 + 2] + bb.z;
      v.w = smem[mloc * 65 + n4 + 3] + bb.w;
      *(float4*)(dst + (size_t)(m0 + mloc) * HS_ + nbg0 * 16 + n4) = v;
    }
  } else {
    // tile [p 64][c 64] stride 65; fuse bias + skip, coalesced float4 along c
    #pragma unroll
    for (int nb = 0; nb < 4; ++nb)
      #pragma unroll
      for (int r = 0; r < 8; ++r)
        smem[(nb * 16 + rowl) * 65 + (wv * 16 + hi * 8 + r)] = acc[nb][r];
    __syncthreads();
    const int p0 = blockIdx.y * 64;
    for (int q = tid; q < 1024; q += 128) {
      const int p  = q >> 6 << 2;  // unused pattern guard (see below)
      const int pp = q >> 4;
      const int c4 = (q & 15) * 4;
      const int pg = p0 + pp;
      (void)p;
      if (pg < P_) {
        const size_t o = (size_t)bB * (P_ * C_) + (size_t)pg * C_ + (c0 + c4);
        const float4 xs = *(const float4*)(xskip + o);
        const float  bb = bias[pg];
        float4 v;
        v.x = smem[pp * 65 + c4 + 0] + bb + xs.x;
        v.y = smem[pp * 65 + c4 + 1] + bb + xs.y;
        v.z = smem[pp * 65 + c4 + 2] + bb + xs.z;
        v.w = smem[pp * 65 + c4 + 3] + bb + xs.w;
        *(float4*)(dst + o) = v;
      }
    }
  }
}

// ---------------------------------------------------------------------------
extern "C" void kernel_launch(void* const* d_in, const int* in_sizes, int n_in,
                              void* d_out, int out_size, void* d_ws, size_t ws_size,
                              hipStream_t stream)
{
  const float* x   = (const float*)d_in[0];
  const float* w1s = (const float*)d_in[1];
  const float* w1b = (const float*)d_in[2];
  const float* b1  = (const float*)d_in[3];
  const float* w2s = (const float*)d_in[4];
  const float* w2b = (const float*)d_in[5];
  const float* b2  = (const float*)d_in[6];
  float* out = (float*)d_out;

  float* h = (float*)d_ws;
  const size_t hBytes = (size_t)M_ * HS_ * sizeof(float);     // 75.5 MB
  const int KCT1 = P_ / 4 + (P_ + 31) / 32;    // 49 + 7  = 56
  const int KCT2 = HS_ / 4 + (HS_ + 31) / 32;  // 96 + 12 = 108
  const int NBP1 = 24, NBP2 = 16;
  const int tot1 = KCT1 * NBP1 * 256;
  const int tot2 = KCT2 * NBP2 * 256;
  unsigned int* wp1 = (unsigned int*)((char*)d_ws + hBytes);
  unsigned int* wp2 = wp1 + tot1;

  kan_pack_w<<<(tot1 + 255) / 256, 256, 0, stream>>>(w1s, w1b, wp1, HS_, P_,  NBP1, P_ / 4,  KCT1);
  kan_pack_w<<<(tot2 + 255) / 256, 256, 0, stream>>>(w2s, w2b, wp2, P_,  HS_, NBP2, HS_ / 4, KCT2);

  kan_gemm<1><<<dim3(M_ / 64, NBP1 / 4), 128, 0, stream>>>(x, wp1, b1, nullptr, h);
  kan_gemm<2><<<dim3(M_ / 64, NBP2 / 4), 128, 0, stream>>>(h, wp2, b2, x, out);
}
